// Model5_13975823581918
// MI455X (gfx1250) — matrix operand order repaired and checked by reading, not compiled, not run on hardware
//
#include <hip/hip_runtime.h>
#include <cstdint>
#include <cstddef>

typedef __bf16 bf16;
typedef __attribute__((ext_vector_type(16))) __bf16 v16bf;
typedef __attribute__((ext_vector_type(8)))  __bf16 v8bf;
typedef __attribute__((ext_vector_type(8)))  float  v8f;

// ---------------------------------------------------------------------------
// WMMA helpers (gfx1250 wave32, v_wmma_f32_16x16x32_bf16)
// ---------------------------------------------------------------------------
__device__ __forceinline__ v8f wmma_bf16(v16bf a, v16bf b, v8f c) {
  return __builtin_amdgcn_wmma_f32_16x16x32_bf16(false, a, false, b, (short)0, c,
                                                 false, false);
}

// Opaque zero (SGPR): adding it to a pointer defeats loop-invariant code
// motion of the weight-fragment loads (which previously caused a 400+ VGPR
// hoist + scratch spill) while preserving pointer provenance so the
// address-space inference still emits global_load_b128 (not flat_load).
__device__ __forceinline__ int opaque_zero() {
  int z = 0;
  asm volatile("" : "+s"(z));
  return z;
}

// A fragment (16x32 bf16) from row-major bf16 row pointer (row = lane&15).
// lanes 0-15: K [0..15]; lanes 16-31: K [16..31].
__device__ __forceinline__ v16bf mk_a(const bf16* p, int hi) {
  v8bf lo8 = *(const v8bf*)(p + (hi ? 16 : 0));
  v8bf hi8 = *(const v8bf*)(p + (hi ? 24 : 8));
  v16bf a;
#pragma unroll
  for (int i = 0; i < 8; ++i) { a[i] = lo8[i]; a[8 + i] = hi8[i]; }
  return a;
}

// B fragment from packed layout Bp[(k>>5)*N*32 + n*32 + (k&31)]:
// lane covers n = lane&15, K = 32*kt + 16*(lane>>4) + e  -> contiguous 32B.
__device__ __forceinline__ v16bf mk_b(const bf16* packBase, int kt, int N, int n,
                                      int hi) {
  return *(const v16bf*)(packBase + ((size_t)kt * N + n) * 32 + 16 * hi);
}

__device__ __forceinline__ float wred(float v) {
#pragma unroll
  for (int m = 16; m >= 1; m >>= 1) v += __shfl_xor(v, m, 32);
  return v;
}

__device__ __forceinline__ float sigf(float x) { return 1.f / (1.f + __expf(-x)); }

// ---------------------------------------------------------------------------
// Embedding gather + cast to bf16, zero-padded Din -> Kpad
// ---------------------------------------------------------------------------
__global__ void embed_gather_k(const int* __restrict__ idx,
                               const float* __restrict__ embed,
                               bf16* __restrict__ out, long rows, int D, int Kpad) {
  long i = (long)blockIdx.x * blockDim.x + threadIdx.x;
  long total = rows * (long)Kpad;
  if (i >= total) return;
  int d = (int)(i % Kpad);
  long r = i / Kpad;
  float v = 0.f;
  if (d < D) v = embed[(long)idx[r] * D + d];
  out[i] = (bf16)v;
}

// ---------------------------------------------------------------------------
// Weight pack: W -> bf16 Bpack[(Kpad/32)][N][32]
//   fromNK=1 : W stored [N][Kdin] (like wih/whh: gi = x @ W.T)
//   fromNK=0 : W stored [Kdin][N]
//   interleave=1 : packed col c corresponds to original output (c%3)*(N/3)+c/3
// ---------------------------------------------------------------------------
__global__ void pack_b_k(const float* __restrict__ W, bf16* __restrict__ out,
                         int Kdin, int Kpad, int N, int fromNK, int interleave) {
  long i = (long)blockIdx.x * blockDim.x + threadIdx.x;
  long total = (long)Kpad * N;
  if (i >= total) return;
  int c = (int)(i % N);
  int k = (int)(i / N);
  int sc = c;
  if (interleave) sc = (c % 3) * (N / 3) + c / 3;
  float v = 0.f;
  if (k < Kdin) v = fromNK ? W[(long)sc * Kdin + k] : W[(long)k * N + sc];
  out[((long)(k >> 5) * N + c) * 32 + (k & 31)] = (bf16)v;
}

// ---------------------------------------------------------------------------
// sentence lengths (count of nonzero tokens)
// ---------------------------------------------------------------------------
__global__ void lens_k(const int* __restrict__ x, int* __restrict__ lens, int S,
                       int T) {
  int s = blockIdx.x * blockDim.x + threadIdx.x;
  if (s >= S) return;
  int c = 0;
  for (int t = 0; t < T; ++t) c += (x[s * T + t] != 0) ? 1 : 0;
  lens[s] = c;
}

// ---------------------------------------------------------------------------
// Fused GRU direction, 8 waves per workgroup.
// Each workgroup owns a 16-row batch tile; wave w owns gate-column groups
// {2w, 2w+1} == GRU units [32w, 32w+32). Per timestep each wave runs its WMMA
// chain (x_t@Wih and h@Whh, gate-interleaved packed B streamed from L2),
// closes its own gates, updates its units in the double-buffered LDS hidden
// state, and keeps the (masked) max-pool in registers. One workgroup barrier
// per timestep.
//   X rows: b*T + t, width Kx (multiple of 32)
//   out: pooled vector, features [dir*256, dir*256+256) at stride outStride
// ---------------------------------------------------------------------------
__global__ __launch_bounds__(256) void gru_fused_k(
    const bf16* __restrict__ X, const bf16* __restrict__ WihP,
    const bf16* __restrict__ WhhP, const float* __restrict__ bih,
    const float* __restrict__ bhh, const int* __restrict__ lens,
    float* __restrict__ outF, bf16* __restrict__ outB, int batch, int T, int Kx,
    int outStride) {
  const int N3 = 768;
  const int dir = blockIdx.y;
  const int sbase = blockIdx.x * 16;
  const int tid = threadIdx.x;
  const int lane = tid & 31;
  const int wv = tid >> 5;  // wave id 0..7
  const int lo = lane & 15, hi = lane >> 4;

  const bf16* wih0 = WihP + (size_t)dir * (Kx / 32) * (size_t)N3 * 32;
  const bf16* whh0 = WhhP + (size_t)dir * 8 * (size_t)N3 * 32;
  const float* bi = bih + dir * N3;
  const float* bh = bhh + dir * N3;

  __shared__ __align__(64) bf16 hb[2][16][256];   // double-buffered hidden (16KB)
  __shared__ __align__(64) float stg[8][16][64];  // per-wave gate staging (32KB)
                                                  // unit uu: [4uu]=r_sum [4uu+1]=z_sum
                                                  //          [4uu+2]=gi_n [4uu+3]=gh_n

  for (int i = tid; i < 2 * 16 * 256; i += 256) (&hb[0][0][0])[i] = (bf16)0.f;

  float rm[16];  // running max for this wave's 16x32 (row,unit) pairs
#pragma unroll
  for (int r = 0; r < 16; ++r) rm[r] = -3.4e38f;

  int rowA = sbase + lo;
  if (rowA >= batch) rowA = batch - 1;  // clamp (rows independent; harmless)
  const int Kx32 = Kx / 32;

  __syncthreads();

  for (int st = 0; st < T; ++st) {
    const int t = dir ? (T - 1 - st) : st;  // input index == output position
    const int cur = st & 1, nxt = cur ^ 1;
    // per-step opaque zero offset: keeps B loads inside the loop (streamed
    // global_load_b128 from L2) instead of hoisted-and-spilled to scratch
    const int zz = opaque_zero();
    const bf16* whh = whh0 + zz;
    const bf16* wih = wih0 + zz;
    const bf16* xrow = X + ((size_t)rowA * T + t) * Kx;

#pragma unroll
    for (int gsub = 0; gsub < 2; ++gsub) {
      const int g = 2 * wv + gsub;
      const int n0 = 48 * g + lo;
      v8f ag0 = {}, ag1 = {}, ag2 = {};
      v8f ai0 = {}, ai1 = {}, ai2 = {};
      // recurrent projection: h (K=256)
      for (int kt = 0; kt < 8; ++kt) {
        v16bf a = mk_a(&hb[cur][lo][kt * 32], hi);
        ag0 = wmma_bf16(a, mk_b(whh, kt, N3, n0, hi), ag0);
        ag1 = wmma_bf16(a, mk_b(whh, kt, N3, n0 + 16, hi), ag1);
        ag2 = wmma_bf16(a, mk_b(whh, kt, N3, n0 + 32, hi), ag2);
      }
      // input projection: x_t (K=Kx)
      for (int kt = 0; kt < Kx32; ++kt) {
        v16bf a = mk_a(xrow + kt * 32, hi);
        ai0 = wmma_bf16(a, mk_b(wih, kt, N3, n0, hi), ai0);
        ai1 = wmma_bf16(a, mk_b(wih, kt, N3, n0 + 16, hi), ai1);
        ai2 = wmma_bf16(a, mk_b(wih, kt, N3, n0 + 32, hi), ai2);
      }
      // spill: packed col cc = 16*j + lo -> unit uu = cc/3, gate = cc%3
#pragma unroll
      for (int j = 0; j < 3; ++j) {
        const int cc = 16 * j + lo;
        const int uu = cc / 3;
        const int gate = cc - 3 * uu;
        const v8f agj = (j == 0) ? ag0 : (j == 1) ? ag1 : ag2;
        const v8f aij = (j == 0) ? ai0 : (j == 1) ? ai1 : ai2;
#pragma unroll
        for (int r = 0; r < 8; ++r) {
          const int rr = r + 8 * hi;
          if (gate < 2) {
            stg[wv][rr][4 * uu + gate] = aij[r] + agj[r];
          } else {
            stg[wv][rr][4 * uu + 2] = aij[r];
            stg[wv][rr][4 * uu + 3] = agj[r];
          }
        }
      }
      // gates for units u = 16g .. 16g+15
#pragma unroll
      for (int i = 0; i < 8; ++i) {
        const int idx = lane + 32 * i;
        const int row = idx & 15, uu = idx >> 4;
        const int u = 16 * g + uu;
        const float gr = stg[wv][row][4 * uu + 0] + bi[u] + bh[u];
        const float gz = stg[wv][row][4 * uu + 1] + bi[256 + u] + bh[256 + u];
        const float r_ = sigf(gr);
        const float z_ = sigf(gz);
        const float n_ = tanhf(stg[wv][row][4 * uu + 2] + bi[512 + u] +
                               r_ * (stg[wv][row][4 * uu + 3] + bh[512 + u]));
        const float ho = (float)hb[cur][row][u];
        const float hn = (1.f - z_) * n_ + z_ * ho;
        hb[nxt][row][u] = (bf16)hn;
        const int srow = sbase + row;
        if (srow < batch && (!lens || t < lens[srow])) {
          const int ridx = gsub * 8 + i;
          rm[ridx] = fmaxf(rm[ridx], hn);
        }
      }
    }
    __syncthreads();
  }

  // write pooled outputs for this wave's units
#pragma unroll
  for (int ridx = 0; ridx < 16; ++ridx) {
    const int gsub = ridx >> 3, i = ridx & 7;
    const int idx = lane + 32 * i;
    const int row = idx & 15, uu = idx >> 4;
    const int u = 32 * wv + 16 * gsub + uu;
    const int srow = sbase + row;
    if (srow < batch) {
      const int len = lens ? lens[srow] : T;
      const float v = (len > 0) ? rm[ridx] : 0.f;
      outF[(size_t)srow * outStride + dir * 256 + u] = v;
      if (outB) outB[(size_t)srow * outStride + dir * 256 + u] = (bf16)v;
    }
  }
}

// ---------------------------------------------------------------------------
// Generic WMMA GEMM: C[M][N] (f32) = A(bf16 [M][K]) x Bpack[(K/32)][N][32]
// one wave per 16x64 tile
// ---------------------------------------------------------------------------
__global__ __launch_bounds__(32) void gemm_wmma_k(const bf16* __restrict__ A,
                                                  const bf16* __restrict__ Bp,
                                                  float* __restrict__ C, int M,
                                                  int N, int K) {
  const int lane = threadIdx.x;
  const int lo = lane & 15, hi = lane >> 4;
  const int mb = blockIdx.x * 16, nb = blockIdx.y * 64;
  v8f c0 = {}, c1 = {}, c2 = {}, c3 = {};
  const bf16* arow = A + (size_t)(mb + lo) * K;
  for (int kt = 0; kt < K / 32; ++kt) {
    v16bf a = mk_a(arow + kt * 32, hi);
    c0 = wmma_bf16(a, mk_b(Bp, kt, N, nb + lo, hi), c0);
    c1 = wmma_bf16(a, mk_b(Bp, kt, N, nb + 16 + lo, hi), c1);
    c2 = wmma_bf16(a, mk_b(Bp, kt, N, nb + 32 + lo, hi), c2);
    c3 = wmma_bf16(a, mk_b(Bp, kt, N, nb + 48 + lo, hi), c3);
  }
#pragma unroll
  for (int r = 0; r < 8; ++r) {
    float* crow = C + (size_t)(mb + r + 8 * hi) * N + nb + lo;
    crow[0] = c0[r];
    crow[16] = c1[r];
    crow[32] = c2[r];
    crow[48] = c3[r];
  }
}

// ---------------------------------------------------------------------------
// Scoring kernels
// ---------------------------------------------------------------------------
__global__ void evsum_k(const float* __restrict__ evv, float* __restrict__ sum,
                        int E) {
  int f = blockIdx.x * blockDim.x + threadIdx.x;
  if (f >= 512) return;
  float s = 0.f;
  for (int e = 0; e < E; ++e) s += evv[(size_t)e * 512 + f];
  sum[f] = s / (float)E;
}

__global__ void salvec_k(const float* __restrict__ W, const float* __restrict__ sum,
                         float* __restrict__ tmp) {
  int d = blockIdx.x * blockDim.x + threadIdx.x;
  if (d >= 512) return;
  float s = 0.f;
  for (int f = 0; f < 512; ++f) s += W[(size_t)d * 512 + f] * sum[f];
  tmp[d] = s;
}

__global__ __launch_bounds__(32) void eventprob_k(
    const float* __restrict__ evv, const float* __restrict__ wc,
    const float* __restrict__ t1, const float* __restrict__ etfs,
    const float* __restrict__ stgt, const float* __restrict__ etf,
    const float* __restrict__ ebias, const float* __restrict__ esent,
    float* __restrict__ out, int eps) {
  const int e = blockIdx.x;
  const int lane = threadIdx.x;
  float pc = 0.f, ps = 0.f;
  for (int f = lane; f < 512; f += 32) {
    const float v = evv[(size_t)e * 512 + f];
    pc += v * wc[f];
    ps += v * t1[f];
  }
  pc = wred(pc);
  ps = wred(ps);
  if (lane == 0)
    out[1024 + e] = pc + etf[0] * etfs[e] + ebias[0] + ps + esent[0] * stgt[e / eps];
}

__global__ __launch_bounds__(32) void sentout_k(
    const float* __restrict__ sv, const float* __restrict__ Vs,
    const float* __restrict__ blog, const float* __restrict__ dv,
    const float* __restrict__ U, const float* __restrict__ evv,
    const float* __restrict__ wsc, const float* __restrict__ dpe,
    const float* __restrict__ spe, const float* __restrict__ wdp,
    const float* __restrict__ wsp, const float* __restrict__ brel,
    const float* __restrict__ epara, const float* __restrict__ sbias,
    float* __restrict__ out) {
  const int s = blockIdx.x;
  const int lane = threadIdx.x;
  const int d = s >> 5, j = s & 31;
  float p = 0.f, p0 = 0.f, p1 = 0.f, p2 = 0.f;
  for (int f = lane; f < 512; f += 32) {
    p += sv[(size_t)s * 512 + f] * wsc[f];
    p += Vs[(size_t)s * 1024 + f] * blog[f];
    p += Vs[(size_t)s * 1024 + 512 + f] * dv[(size_t)d * 512 + f];
    const float uf = U[(size_t)s * 512 + f];
    p0 += uf * evv[((size_t)(3 * s + 0)) * 512 + f];
    p1 += uf * evv[((size_t)(3 * s + 1)) * 512 + f];
    p2 += uf * evv[((size_t)(3 * s + 2)) * 512 + f];
  }
  const int dpos = (d * 40) / 32, spos = (j * 30) / 32;
  for (int q = lane; q < 50; q += 32)
    p += dpe[dpos * 50 + q] * wdp[q] + spe[spos * 50 + q] * wsp[q];
  p = wred(p);
  p0 = wred(p0);
  p1 = wred(p1);
  p2 = wred(p2);
  if (lane == 0) {
    const float b = brel[0];
    const float er = (p0 + b) * out[1024 + 3 * s + 0] +
                     (p1 + b) * out[1024 + 3 * s + 1] +
                     (p2 + b) * out[1024 + 3 * s + 2];
    out[s] = p + epara[0] * er + sbias[0];
  }
}

// ---------------------------------------------------------------------------
// Host launcher
// ---------------------------------------------------------------------------
extern "C" void kernel_launch(void* const* d_in, const int* in_sizes, int n_in,
                              void* d_out, int out_size, void* d_ws, size_t ws_size,
                              hipStream_t stream) {
  (void)n_in;
  (void)out_size;
  (void)ws_size;
  (void)in_sizes;
  // Problem constants (from setup_inputs)
  const int S = 1024, T = 32, D = 300, DPAD = 320, H2 = 512, ND = 32, E = 3072,
            EPS = 3, ET = 4;
  const int N3 = 768;

  const int* x = (const int*)d_in[0];
  const int* events = (const int*)d_in[1];
  const float* event_tfs = (const float*)d_in[5];
  const float* sent_targets = (const float*)d_in[7];
  const float* embed = (const float*)d_in[8];
  const float* word_wih = (const float*)d_in[9];
  const float* word_whh = (const float*)d_in[10];
  const float* word_bih = (const float*)d_in[11];
  const float* word_bhh = (const float*)d_in[12];
  const float* sent_wih = (const float*)d_in[13];
  const float* sent_whh = (const float*)d_in[14];
  const float* sent_bih = (const float*)d_in[15];
  const float* sent_bhh = (const float*)d_in[16];
  const float* doc_wih = (const float*)d_in[17];
  const float* doc_whh = (const float*)d_in[18];
  const float* doc_bih = (const float*)d_in[19];
  const float* doc_bhh = (const float*)d_in[20];
  const float* ev_wih = (const float*)d_in[21];
  const float* ev_whh = (const float*)d_in[22];
  const float* ev_bih = (const float*)d_in[23];
  const float* ev_bhh = (const float*)d_in[24];
  const float* doc_pos_embed = (const float*)d_in[25];
  const float* sent_pos_embed = (const float*)d_in[26];
  const float* w_event_content = (const float*)d_in[27];
  const float* W_event_sal = (const float*)d_in[28];
  const float* event_tf = (const float*)d_in[29];
  const float* event_sent = (const float*)d_in[30];
  const float* event_bias = (const float*)d_in[31];
  const float* w_sent_content = (const float*)d_in[32];
  const float* W_sent_sal = (const float*)d_in[33];
  const float* w_sent_doc_pos = (const float*)d_in[34];
  const float* w_sent_pos = (const float*)d_in[35];
  const float* W_event_rel = (const float*)d_in[36];
  const float* b_event_rel = (const float*)d_in[37];
  const float* event_para = (const float*)d_in[38];
  const float* sent_bias_p = (const float*)d_in[40];

  float* out = (float*)d_out;

  // workspace carve-up
  char* w = (char*)d_ws;
  size_t off = 0;
  auto alloc = [&](size_t bytes) -> void* {
    void* p = w + off;
    off = (off + bytes + 255) & ~(size_t)255;
    return p;
  };
  bf16* Xw = (bf16*)alloc((size_t)S * T * DPAD * 2);
  bf16* Evw = (bf16*)alloc((size_t)E * ET * DPAD * 2);
  bf16* pWordWih = (bf16*)alloc((size_t)2 * DPAD * N3 * 2);
  bf16* pWordWhh = (bf16*)alloc((size_t)2 * 256 * N3 * 2);
  bf16* pSentWih = (bf16*)alloc((size_t)2 * 512 * N3 * 2);
  bf16* pSentWhh = (bf16*)alloc((size_t)2 * 256 * N3 * 2);
  bf16* pDocWih = (bf16*)alloc((size_t)2 * 512 * N3 * 2);
  bf16* pDocWhh = (bf16*)alloc((size_t)2 * 256 * N3 * 2);
  bf16* pEvWih = (bf16*)alloc((size_t)2 * DPAD * N3 * 2);
  bf16* pEvWhh = (bf16*)alloc((size_t)2 * 256 * N3 * 2);
  bf16* pWrel = (bf16*)alloc((size_t)512 * 512 * 2);
  bf16* pWsal = (bf16*)alloc((size_t)512 * 1024 * 2);
  float* sentVecF = (float*)alloc((size_t)S * H2 * 4);
  bf16* sentVecB = (bf16*)alloc((size_t)S * H2 * 2);
  float* docVecF = (float*)alloc((size_t)ND * H2 * 4);
  bf16* docVecB = (bf16*)alloc((size_t)ND * H2 * 2);
  float* blogF = (float*)alloc((size_t)H2 * 4);
  float* evVecF = (float*)alloc((size_t)E * H2 * 4);
  bf16* evVecB = (bf16*)alloc((size_t)E * H2 * 2);
  float* U = (float*)alloc((size_t)S * 512 * 4);
  float* Vsal = (float*)alloc((size_t)S * 1024 * 4);
  float* evSum = (float*)alloc(512 * 4);
  float* tmp1 = (float*)alloc(512 * 4);
  int* sentLens = (int*)alloc((size_t)S * 4);

  // 1) embeddings -> bf16 (padded to 320)
  {
    long tot = (long)S * T * DPAD;
    embed_gather_k<<<dim3((unsigned)((tot + 255) / 256)), 256, 0, stream>>>(
        x, embed, Xw, (long)S * T, D, DPAD);
    long tot2 = (long)E * ET * DPAD;
    embed_gather_k<<<dim3((unsigned)((tot2 + 255) / 256)), 256, 0, stream>>>(
        events, embed, Evw, (long)E * ET, D, DPAD);
  }
  lens_k<<<dim3((S + 255) / 256), 256, 0, stream>>>(x, sentLens, S, T);

  // 2) pack weights (per direction), gate-interleaved for GRUs
  auto pack = [&](const float* W, bf16* o, int Kdin, int Kpad, int N, int fromNK,
                  int il) {
    long tot = (long)Kpad * N;
    pack_b_k<<<dim3((unsigned)((tot + 255) / 256)), 256, 0, stream>>>(
        W, o, Kdin, Kpad, N, fromNK, il);
  };
  for (int dir = 0; dir < 2; ++dir) {
    pack(word_wih + (size_t)dir * N3 * D, pWordWih + (size_t)dir * DPAD * N3, D,
         DPAD, N3, 1, 1);
    pack(word_whh + (size_t)dir * N3 * 256, pWordWhh + (size_t)dir * 256 * N3, 256,
         256, N3, 1, 1);
    pack(sent_wih + (size_t)dir * N3 * 512, pSentWih + (size_t)dir * 512 * N3, 512,
         512, N3, 1, 1);
    pack(sent_whh + (size_t)dir * N3 * 256, pSentWhh + (size_t)dir * 256 * N3, 256,
         256, N3, 1, 1);
    pack(doc_wih + (size_t)dir * N3 * 512, pDocWih + (size_t)dir * 512 * N3, 512,
         512, N3, 1, 1);
    pack(doc_whh + (size_t)dir * N3 * 256, pDocWhh + (size_t)dir * 256 * N3, 256,
         256, N3, 1, 1);
    pack(ev_wih + (size_t)dir * N3 * D, pEvWih + (size_t)dir * DPAD * N3, D, DPAD,
         N3, 1, 1);
    pack(ev_whh + (size_t)dir * N3 * 256, pEvWhh + (size_t)dir * 256 * N3, 256, 256,
         N3, 1, 1);
  }
  pack(W_event_rel, pWrel, 512, 512, 512, 0, 0);
  pack(W_sent_sal, pWsal, 512, 512, 1024, 0, 0);

  // 3) fused BiGRU levels (grid.y = direction, 8 waves per workgroup)
  // word: batch=1024, T=32, masked max pool -> sent_vec
  gru_fused_k<<<dim3(S / 16, 2), 256, 0, stream>>>(Xw, pWordWih, pWordWhh, word_bih,
                                                   word_bhh, sentLens, sentVecF,
                                                   sentVecB, S, T, DPAD, H2);
  // event: batch=3072, T=4, full max pool -> event_vec
  gru_fused_k<<<dim3(E / 16, 2), 256, 0, stream>>>(Evw, pEvWih, pEvWhh, ev_bih,
                                                   ev_bhh, nullptr, evVecF, evVecB,
                                                   E, ET, DPAD, H2);
  // sentence-level: batch=32 docs, T=32 sentences -> doc_vec
  gru_fused_k<<<dim3(ND / 16, 2), 256, 0, stream>>>(sentVecB, pSentWih, pSentWhh,
                                                    sent_bih, sent_bhh, nullptr,
                                                    docVecF, docVecB, ND, 32, 512,
                                                    H2);
  // doc-level: batch=1, T=32 docs -> blog_vec
  gru_fused_k<<<dim3(1, 2), 256, 0, stream>>>(docVecB, pDocWih, pDocWhh, doc_bih,
                                              doc_bhh, nullptr, blogF, nullptr, 1,
                                              ND, 512, H2);

  // 4) scoring GEMMs (WMMA): U = SV @ W_event_rel ; Vsal = SV @ W_sent_sal
  gemm_wmma_k<<<dim3(S / 16, 512 / 64), 32, 0, stream>>>(sentVecB, pWrel, U, S, 512,
                                                         512);
  gemm_wmma_k<<<dim3(S / 16, 1024 / 64), 32, 0, stream>>>(sentVecB, pWsal, Vsal, S,
                                                          1024, 512);

  // 5) event scores -> out[1024 .. 4096)
  evsum_k<<<dim3(2), 256, 0, stream>>>(evVecF, evSum, E);
  salvec_k<<<dim3(2), 256, 0, stream>>>(W_event_sal, evSum, tmp1);
  eventprob_k<<<dim3(E), 32, 0, stream>>>(evVecF, w_event_content, tmp1, event_tfs,
                                          sent_targets, event_tf, event_bias,
                                          event_sent, out, EPS);

  // 6) sentence scores -> out[0 .. 1024)
  sentout_k<<<dim3(S), 32, 0, stream>>>(sentVecF, Vsal, blogF, docVecF, U, evVecF,
                                        w_sent_content, doc_pos_embed,
                                        sent_pos_embed, w_sent_doc_pos, w_sent_pos,
                                        b_event_rel, event_para, sent_bias_p, out);
}
